// AttentionGNN_Encoder_81389630259525
// MI455X (gfx1250) — matrix operand order repaired and checked by reading, not compiled, not run on hardware
//
#include <hip/hip_runtime.h>
#include <hip/hip_bf16.h>
#include <float.h>
#include <limits.h>

// Problem constants (from reference)
#define NUQ   2560
#define NIQ   3584
#define NT    6144          // NU + NI
#define EE    64            // embedding dim
#define DQK   256           // 4*E
#define NNZ   200000
#define TOPK  5

typedef __attribute__((ext_vector_type(16))) _Float16 v16h;
typedef __attribute__((ext_vector_type(8)))  _Float16 v8h;
typedef __attribute__((ext_vector_type(8)))  float    v8f;

// ---------------------------------------------------------------------------
// WMMA helpers (layouts per CDNA5 ISA 7.12.2, wave32)
// ---------------------------------------------------------------------------
__device__ __forceinline__ v8f wmma_f16(v16h a, v16h b, v8f c) {
  return __builtin_amdgcn_wmma_f32_16x16x32_f16(
      /*neg_a=*/false, a, /*neg_b=*/false, b,
      /*c_mod=*/(short)0, c, /*reuse_a=*/false, /*reuse_b=*/false);
}

// A fragment: 16x32 f16 tile of row-major X (ld = EE), rows row0..row0+15,
// cols k0..k0+31.  Lane layout: lanes 0-15 K={0..15}, lanes 16-31
// K={16..31} relative to k0 (one contiguous 16-half chunk per lane).
__device__ __forceinline__ v16h load_afrag(const _Float16* X, int row0, int k0, int lane) {
  const int hf = lane >> 4, r = lane & 15;
  return *(const v16h*)(X + (row0 + r) * EE + k0 + 16 * hf);
}

// B fragment: B[k,n] = X[col0+n, k0+k] for row-major X (ld = EE); per-lane this
// is 16 contiguous halfs of row (col0 + lane&15) starting at k0 + 16*(lane>>4).
__device__ __forceinline__ v16h load_bfrag(const _Float16* X, int col0, int k0, int lane) {
  const int hf = lane >> 4, r = lane & 15;
  return *(const v16h*)(X + (col0 + r) * EE + k0 + 16 * hf);
}

// ---------------------------------------------------------------------------
// Small utility kernels
// ---------------------------------------------------------------------------
__global__ void k_zero_i32(int* p, int n) {
  int i = blockIdx.x * blockDim.x + threadIdx.x;
  if (i < n) p[i] = 0;
}

__global__ void k_concat_ego0(const float* u, const float* it, float* ego0) {
  int i = blockIdx.x * blockDim.x + threadIdx.x;
  const int tot = NT * EE;
  if (i < tot) ego0[i] = (i < NUQ * EE) ? u[i] : it[i - NUQ * EE];
}

__global__ void k_mean(const float* a, const float* b, float* out) {
  int i = blockIdx.x * blockDim.x + threadIdx.x;
  if (i < NT * EE) out[i] = 0.5f * (a[i] + b[i]);
}

__global__ void k_f32_to_f16(const float* s, _Float16* d, int n) {
  int i = blockIdx.x * blockDim.x + threadIdx.x;
  if (i < n) d[i] = (_Float16)s[i];
}

// ---------------------------------------------------------------------------
// CSR build (duplicates preserved; addition is commutative so this matches the
// reference's scatter-add densification semantics)
// ---------------------------------------------------------------------------
__global__ void k_hist(const int* rows, int* cnt, int nnz) {
  int e = blockIdx.x * blockDim.x + threadIdx.x;
  if (e < nnz) atomicAdd(&cnt[rows[e]], 1);
}

__global__ void k_scan(const int* cnt, int* off, int* cursor, int n) {
  if (blockIdx.x == 0 && threadIdx.x == 0) {
    int s = 0;
    for (int i = 0; i < n; ++i) { off[i] = s; cursor[i] = s; s += cnt[i]; }
    off[n] = s;
  }
}

__global__ void k_scatter(const int* rows, const int* cols, const float* vals,
                          int* cursor, int* colS, float* valS, int nnz) {
  int e = blockIdx.x * blockDim.x + threadIdx.x;
  if (e < nnz) {
    int r = rows[e];
    int p = atomicAdd(&cursor[r], 1);
    colS[p] = cols[e];
    valS[p] = vals[e];
  }
}

// ego_out[r,:] = sum_nbrs v * ego_in[c,:]   (one block of EE threads per row)
__global__ void k_spmm(const int* off, const int* colS, const float* valS,
                       const float* Xin, float* Xout) {
  int r = blockIdx.x;
  int t = threadIdx.x;            // 0..63
  int s = off[r], e = off[r + 1];
  float acc = 0.f;
  for (int k = s; k < e; ++k) acc += valS[k] * Xin[colS[k] * EE + t];
  Xout[r * EE + t] = acc;
}

// ---------------------------------------------------------------------------
// sim = ego2 @ ego2^T   (WMMA f16->f32, K=64 = two k-steps)
// grid: (NT/16, NT/(8*16)), block 256 (8 waves, one 16x16 tile each)
// ---------------------------------------------------------------------------
__global__ void k_sim_gemm(const _Float16* egoH, float* simA) {
  const int wave = threadIdx.x >> 5;
  const int lane = threadIdx.x & 31;
  const int i0 = blockIdx.x * 16;
  const int j0 = (blockIdx.y * 8 + wave) * 16;
  v8f acc = {};
  acc = wmma_f16(load_afrag(egoH, i0, 0,  lane), load_bfrag(egoH, j0, 0,  lane), acc);
  acc = wmma_f16(load_afrag(egoH, i0, 32, lane), load_bfrag(egoH, j0, 32, lane), acc);
  const int hf = lane >> 4, r = lane & 15;
#pragma unroll
  for (int v = 0; v < 8; ++v) {
    int m = i0 + v + 8 * hf;
    simA[(size_t)m * NT + j0 + r] = acc[v];
  }
}

// simB[r,j] = simA[r,j] + 0.5 * sum_nbrs v * simA[c,j]
// grid: (NT, NT/256), block 256
__global__ void k_blend(const int* off, const int* colS, const float* valS,
                        const float* simA, float* simB) {
  int r = blockIdx.x;
  int j = blockIdx.y * 256 + threadIdx.x;
  float acc = simA[(size_t)r * NT + j];
  int s = off[r], e = off[r + 1];
  for (int k = s; k < e; ++k)
    acc += 0.5f * valS[k] * simA[(size_t)colS[k] * NT + j];
  simB[(size_t)r * NT + j] = acc;
}

// ---------------------------------------------------------------------------
// Top-K (K=5) per row; value desc, index asc tie-break
// ---------------------------------------------------------------------------
__device__ __forceinline__ void ins5(float v, int j, float* bv, int* bi) {
#pragma unroll
  for (int s = 0; s < 5; ++s) {
    bool better = (v > bv[s]) || (v == bv[s] && j < bi[s]);
    if (better) {
      for (int t = 4; t > s; --t) { bv[t] = bv[t - 1]; bi[t] = bi[t - 1]; }
      bv[s] = v; bi[s] = j;
      return;
    }
  }
}

__global__ void k_topk(const float* simB, int* topk_idx) {
  __shared__ float sv[256 * TOPK];
  __shared__ int   si[256 * TOPK];
  int row = blockIdx.x;
  const float* p = simB + (size_t)row * NT;
  float bv[5]; int bi[5];
#pragma unroll
  for (int s = 0; s < 5; ++s) { bv[s] = -FLT_MAX; bi[s] = INT_MAX; }
  for (int j = threadIdx.x; j < NT; j += 256) ins5(p[j], j, bv, bi);
#pragma unroll
  for (int s = 0; s < 5; ++s) { sv[threadIdx.x * 5 + s] = bv[s]; si[threadIdx.x * 5 + s] = bi[s]; }
  __syncthreads();
  if (threadIdx.x == 0) {
    float gv[5]; int gi[5];
#pragma unroll
    for (int s = 0; s < 5; ++s) { gv[s] = -FLT_MAX; gi[s] = INT_MAX; }
    for (int t = 0; t < 256 * 5; ++t)
      if (si[t] != INT_MAX) ins5(sv[t], si[t], gv, gi);
#pragma unroll
    for (int s = 0; s < 5; ++s) topk_idx[row * TOPK + s] = gi[s];
  }
}

// Gather sampled neighbor embeddings into f16 matrix S [NT*TOPK, EE]
__global__ void k_gather(const int* topk_idx, const float* ego2, _Float16* SH) {
  int s = blockIdx.x;                 // 0 .. NT*TOPK-1
  int t = threadIdx.x;                // 0 .. 63
  int src = topk_idx[s];
  SH[s * EE + t] = (_Float16)ego2[src * EE + t];
}

// ---------------------------------------------------------------------------
// Y[M x 256] = X[M x 64] @ W^T + bias   (W is [256 x 64] row-major, f16)
// grid: (M/16, 2), block 256 (8 waves => 8 n-tiles per block)
// ---------------------------------------------------------------------------
__global__ void k_lin_gemm(const _Float16* X, const _Float16* W, const float* bias,
                           float* Y) {
  const int wave = threadIdx.x >> 5;
  const int lane = threadIdx.x & 31;
  const int i0 = blockIdx.x * 16;
  const int n0 = (blockIdx.y * 8 + wave) * 16;
  v8f acc = {};
  v16h a0 = load_afrag(X, i0, 0,  lane);
  v16h a1 = load_afrag(X, i0, 32, lane);
  acc = wmma_f16(a0, load_bfrag(W, n0, 0,  lane), acc);
  acc = wmma_f16(a1, load_bfrag(W, n0, 32, lane), acc);
  const int hf = lane >> 4, r = lane & 15;
  float b = bias[n0 + r];
#pragma unroll
  for (int v = 0; v < 8; ++v)
    Y[(size_t)(i0 + v + 8 * hf) * DQK + n0 + r] = acc[v] + b;
}

// Fused K & V projection over gathered samples (shares the A fragments)
__global__ void k_kv_gemm(const _Float16* S, const _Float16* Wk, const float* bk,
                          float* SK, const _Float16* Wv, const float* bv, float* SV) {
  const int wave = threadIdx.x >> 5;
  const int lane = threadIdx.x & 31;
  const int i0 = blockIdx.x * 16;
  const int n0 = (blockIdx.y * 8 + wave) * 16;
  v8f ak = {}, av = {};
  v16h a0 = load_afrag(S, i0, 0,  lane);
  v16h a1 = load_afrag(S, i0, 32, lane);
  ak = wmma_f16(a0, load_bfrag(Wk, n0, 0,  lane), ak);
  ak = wmma_f16(a1, load_bfrag(Wk, n0, 32, lane), ak);
  av = wmma_f16(a0, load_bfrag(Wv, n0, 0,  lane), av);
  av = wmma_f16(a1, load_bfrag(Wv, n0, 32, lane), av);
  const int hf = lane >> 4, r = lane & 15;
  float bbk = bk[n0 + r];
  float bbv = bv[n0 + r];
#pragma unroll
  for (int v = 0; v < 8; ++v) {
    size_t o = (size_t)(i0 + v + 8 * hf) * DQK + n0 + r;
    SK[o] = ak[v] + bbk;
    SV[o] = av[v] + bbv;
  }
}

// ---------------------------------------------------------------------------
// scores = Q.K/16, softmax over 5, att_out = att.V    (one wave32 per row)
// ---------------------------------------------------------------------------
__global__ void k_attn(const float* Q, const float* SK, const float* SV, float* att_out) {
  int row  = blockIdx.x * 8 + (threadIdx.x >> 5);
  int lane = threadIdx.x & 31;
  const float* qp = Q + (size_t)row * DQK;
  float sc[TOPK];
#pragma unroll
  for (int k = 0; k < TOPK; ++k) {
    const float* kp = SK + (size_t)(row * TOPK + k) * DQK;
    float part = 0.f;
    for (int d = lane; d < DQK; d += 32) part += qp[d] * kp[d];
#pragma unroll
    for (int off = 16; off > 0; off >>= 1) part += __shfl_xor(part, off, 32);
    sc[k] = part * (1.0f / 16.0f);   // 1/sqrt(DQK)
  }
  float mx = sc[0];
#pragma unroll
  for (int k = 1; k < TOPK; ++k) mx = fmaxf(mx, sc[k]);
  float se = 0.f;
#pragma unroll
  for (int k = 0; k < TOPK; ++k) { sc[k] = __expf(sc[k] - mx); se += sc[k]; }
  float inv = 1.0f / se;
#pragma unroll
  for (int k = 0; k < TOPK; ++k) sc[k] *= inv;
  for (int d = lane; d < DQK; d += 32) {
    float acc = 0.f;
#pragma unroll
    for (int k = 0; k < TOPK; ++k)
      acc += sc[k] * SV[(size_t)(row * TOPK + k) * DQK + d];
    att_out[(size_t)row * DQK + d] = acc;
  }
}

// ---------------------------------------------------------------------------
// Host-side orchestration
// ---------------------------------------------------------------------------
extern "C" void kernel_launch(void* const* d_in, const int* in_sizes, int n_in,
                              void* d_out, int out_size, void* d_ws, size_t ws_size,
                              hipStream_t stream) {
  const float* user_emb  = (const float*)d_in[0];
  const float* item_emb  = (const float*)d_in[1];
  const int*   norm_rows = (const int*)  d_in[2];
  const int*   norm_cols = (const int*)  d_in[3];
  const float* norm_vals = (const float*)d_in[4];
  const int*   adj_rows  = (const int*)  d_in[5];
  const int*   adj_cols  = (const int*)  d_in[6];
  const float* adj_vals  = (const float*)d_in[7];
  const float* Wq = (const float*)d_in[8];
  const float* bq = (const float*)d_in[9];
  const float* Wk = (const float*)d_in[10];
  const float* bk = (const float*)d_in[11];
  const float* Wv = (const float*)d_in[12];
  const float* bv = (const float*)d_in[13];

  float* out_mean = (float*)d_out;                 // [NT, EE]
  float* out_att  = (float*)d_out + NT * EE;       // [NT, DQK]

  // Bump allocator over d_ws
  char* wp = (char*)d_ws;
  auto alloc = [&](size_t bytes) -> void* {
    void* r = (void*)wp;
    wp += (bytes + 255) & ~(size_t)255;
    return r;
  };

  float*    simA   = (float*)   alloc((size_t)NT * NT * 4);
  float*    simB   = (float*)   alloc((size_t)NT * NT * 4);
  float*    ego0   = (float*)   alloc((size_t)NT * EE * 4);
  float*    ego1   = (float*)   alloc((size_t)NT * EE * 4);
  float*    ego2   = (float*)   alloc((size_t)NT * EE * 4);
  _Float16* ego2H  = (_Float16*)alloc((size_t)NT * EE * 2);
  _Float16* WqH    = (_Float16*)alloc((size_t)DQK * EE * 2);
  _Float16* WkH    = (_Float16*)alloc((size_t)DQK * EE * 2);
  _Float16* WvH    = (_Float16*)alloc((size_t)DQK * EE * 2);
  float*    Qm     = (float*)   alloc((size_t)NT * DQK * 4);
  _Float16* SH     = (_Float16*)alloc((size_t)NT * TOPK * EE * 2);
  float*    SKm    = (float*)   alloc((size_t)NT * TOPK * DQK * 4);
  float*    SVm    = (float*)   alloc((size_t)NT * TOPK * DQK * 4);
  int*      tkidx  = (int*)     alloc((size_t)NT * TOPK * 4);
  // CSR for norm adjacency
  int*   cntN = (int*)  alloc((NT + 1) * 4);
  int*   offN = (int*)  alloc((NT + 1) * 4);
  int*   curN = (int*)  alloc((NT + 1) * 4);
  int*   colN = (int*)  alloc((size_t)NNZ * 4);
  float* valN = (float*)alloc((size_t)NNZ * 4);
  // CSR for plain adjacency
  int*   cntA = (int*)  alloc((NT + 1) * 4);
  int*   offA = (int*)  alloc((NT + 1) * 4);
  int*   curA = (int*)  alloc((NT + 1) * 4);
  int*   colA = (int*)  alloc((size_t)NNZ * 4);
  float* valA = (float*)alloc((size_t)NNZ * 4);
  (void)ws_size; (void)n_in; (void)in_sizes; (void)out_size;

  const int TB = 256;
  const int gNNZ = (NNZ + TB - 1) / TB;

  // --- build CSRs (A constant across layers; built once) ---
  k_zero_i32<<<(NT + TB - 1) / TB, TB, 0, stream>>>(cntN, NT);
  k_zero_i32<<<(NT + TB - 1) / TB, TB, 0, stream>>>(cntA, NT);
  k_hist<<<gNNZ, TB, 0, stream>>>(norm_rows, cntN, NNZ);
  k_hist<<<gNNZ, TB, 0, stream>>>(adj_rows,  cntA, NNZ);
  k_scan<<<1, 1, 0, stream>>>(cntN, offN, curN, NT);
  k_scan<<<1, 1, 0, stream>>>(cntA, offA, curA, NT);
  k_scatter<<<gNNZ, TB, 0, stream>>>(norm_rows, norm_cols, norm_vals, curN, colN, valN, NNZ);
  k_scatter<<<gNNZ, TB, 0, stream>>>(adj_rows,  adj_cols,  adj_vals,  curA, colA, valA, NNZ);

  // --- ego propagation ---
  k_concat_ego0<<<(NT * EE + TB - 1) / TB, TB, 0, stream>>>(user_emb, item_emb, ego0);
  k_spmm<<<NT, EE, 0, stream>>>(offN, colN, valN, ego0, ego1);
  k_mean<<<(NT * EE + TB - 1) / TB, TB, 0, stream>>>(ego0, ego1, out_mean);
  k_spmm<<<NT, EE, 0, stream>>>(offN, colN, valN, ego1, ego2);

  // --- f16 conversions for WMMA ---
  k_f32_to_f16<<<(NT * EE + TB - 1) / TB, TB, 0, stream>>>(ego2, ego2H, NT * EE);
  k_f32_to_f16<<<(DQK * EE + TB - 1) / TB, TB, 0, stream>>>(Wq, WqH, DQK * EE);
  k_f32_to_f16<<<(DQK * EE + TB - 1) / TB, TB, 0, stream>>>(Wk, WkH, DQK * EE);
  k_f32_to_f16<<<(DQK * EE + TB - 1) / TB, TB, 0, stream>>>(Wv, WvH, DQK * EE);

  // --- sim = ego2 @ ego2^T (WMMA) ---
  dim3 gSim(NT / 16, NT / 128);         // (384, 48), 8 waves/block
  k_sim_gemm<<<gSim, TB, 0, stream>>>(ego2H, simA);

  // --- simB = simA + 0.5 * A @ simA (sparse) ---
  dim3 gBlend(NT, NT / 256);
  k_blend<<<gBlend, TB, 0, stream>>>(offA, colA, valA, simA, simB);

  // --- top-5 per row ---
  k_topk<<<NT, TB, 0, stream>>>(simB, tkidx);
  k_gather<<<NT * TOPK, EE, 0, stream>>>(tkidx, ego2, SH);

  // --- Q/K/V projections (WMMA) ---
  dim3 gQ(NT / 16, DQK / 128);          // (384, 2)
  k_lin_gemm<<<gQ, TB, 0, stream>>>(ego2H, WqH, bq, Qm);
  dim3 gKV(NT * TOPK / 16, DQK / 128);  // (1920, 2)
  k_kv_gemm<<<gKV, TB, 0, stream>>>(SH, WkH, bk, SKm, WvH, bv, SVm);

  // --- attention: scores, softmax(5), weighted sum ---
  k_attn<<<NT / 8, TB, 0, stream>>>(Qm, SKm, SVm, out_att);
}
